// MultiHeadGATLayer_59691455480216
// MI455X (gfx1250) — matrix operand order repaired and checked by reading, not compiled, not run on hardware
//
#include <hip/hip_runtime.h>
#include <hip/hip_bf16.h>

// ---------------------------------------------------------------------------
// MultiHeadGAT layer for MI455X (gfx1250, wave32).
// Projection GEMM: v_wmma_f32_16x16x32_bf16 with B staged in LDS via the
// gfx1250 async global->LDS path; B tiles batch-loaded per K-step so WMMAs
// issue back-to-back. Rest is the bandwidth/atomic path.
// ---------------------------------------------------------------------------

#define NN   25000
#define NE   400000
#define IND  512
#define OUTD 128
#define EDD  64
#define NH   5
#define MT   ((NN + 15) / 16)    // 1563 M-tiles
#define MB   ((MT + 7) / 8)      // 196 tile-blocks (8 tiles per workgroup)
#define NEG_SLOPE 0.2f
#define EPS_DEN   1e-16f

typedef __attribute__((ext_vector_type(16))) __bf16 v16bf;
typedef __attribute__((ext_vector_type(8)))  __bf16 v8bf;
typedef __attribute__((ext_vector_type(8)))  float  v8f;
typedef int v4i_ty __attribute__((vector_size(16)));

#if defined(__has_builtin)
#if __has_builtin(__builtin_amdgcn_global_load_async_to_lds_b128) && \
    __has_builtin(__builtin_amdgcn_s_wait_asynccnt)
#define USE_ASYNC_LDS 1
#endif
#endif
#ifndef USE_ASYNC_LDS
#define USE_ASYNC_LDS 0
#endif

// ---------------------------------------------------------------- helpers
__device__ __forceinline__ void atomicMaxF(float* addr, float v) {
    // order-preserving float max via int/uint atomics; valid with -inf init
    if (v >= 0.0f) atomicMax((int*)addr, __float_as_int(v));
    else           atomicMin((unsigned int*)addr, __float_as_uint(v));
}

// ---------------------------------------------------------------- prep
__global__ void k_pack_x(const float* __restrict__ x, __bf16* __restrict__ xb) {
    int i = blockIdx.x * blockDim.x + threadIdx.x;
    if (i >= NN * IND / 4) return;
    float4 v = ((const float4*)x)[i];
    __bf16* p = xb + (size_t)i * 4;
    p[0] = (__bf16)v.x; p[1] = (__bf16)v.y; p[2] = (__bf16)v.z; p[3] = (__bf16)v.w;
}

// Pack W[h][k][o] (fp32) into WMMA B-operand layout:
//   wpk[((h*16+kk)*8+t)*32*16 + lane*16 + i] = W[h][kk*32 + (lane>=16?16:0)+i][t*16 + (lane&15)]
__global__ void k_pack_w(const float* __restrict__ W, __bf16* __restrict__ wpk) {
    int idx = blockIdx.x * blockDim.x + threadIdx.x;
    if (idx >= NH * 16 * 8 * 32 * 16) return;
    int i = idx;
    int half = i & 15;  i >>= 4;
    int lane = i & 31;  i >>= 5;
    int t    = i & 7;   i >>= 3;
    int kk   = i & 15;  i >>= 4;
    int h    = i;
    int k = kk * 32 + ((lane >= 16) ? 16 : 0) + half;
    int o = t * 16 + (lane & 15);
    wpk[idx] = (__bf16)W[((size_t)h * IND + k) * OUTD + o];
}

// v_edge[h][d] = sum_o W_edge[h][d][o] * att_edge[h][o]
__global__ void k_vedge(const float* __restrict__ We, const float* __restrict__ ae,
                        float* __restrict__ ve) {
    int t = threadIdx.x;
    if (t >= NH * EDD) return;
    int h = t / EDD, d = t % EDD;
    float s = 0.0f;
    for (int o = 0; o < OUTD; ++o)
        s += We[((size_t)h * EDD + d) * OUTD + o] * ae[h * OUTD + o];
    ve[h * EDD + d] = s;
}

// out = bias (broadcast), denom = 0, seg_max = -inf  (fresh every launch)
__global__ void k_init(float* __restrict__ out, const float* __restrict__ bias,
                       float* __restrict__ denom, float* __restrict__ segmax) {
    int i = blockIdx.x * blockDim.x + threadIdx.x;
    if (i < NN * NH * OUTD) out[i] = bias[i % (NH * OUTD)];
    if (i < NN * NH) { denom[i] = 0.0f; segmax[i] = -__builtin_huge_valf(); }
}

// ---------------------------------------------------------------- WMMA GEMM
// Workgroup = 8 waves = one head x 8 M-tiles. Packed B for the head is staged
// into LDS in two 64KB halves via the async global->LDS path; each wave then
// runs 16x16x32 bf16 WMMAs with all 8 B tiles of a K-step preloaded so the
// matrix ops issue back-to-back behind a single dscnt wait.
__global__ __launch_bounds__(256) void k_gemm(const __bf16* __restrict__ xb,
                                              const __bf16* __restrict__ wpk,
                                              float* __restrict__ xp) {
    __shared__ __attribute__((aligned(32))) __bf16 lds_b[8 * 8 * 32 * 16]; // 64 KB

    int waveid = threadIdx.x >> 5;
    int lane   = threadIdx.x & 31;
    int h      = blockIdx.x / MB;
    int mblock = blockIdx.x % MB;
    int mtile  = mblock * 8 + waveid;
    bool active = (mtile < MT);

    int row  = (active ? mtile : 0) * 16 + (lane & 15);
    int arow = row < NN ? row : NN - 1;   // clamp loads; stores are guarded
    int abase = (lane < 16) ? 0 : 16;     // contiguous K layout (matches packed B)
    const __bf16* aptr = xb + (size_t)arow * IND + abase;

    v8f acc[8];
#pragma unroll
    for (int t = 0; t < 8; ++t) acc[t] = {};

    for (int hh = 0; hh < 2; ++hh) {
        // ---- stage 64KB of packed B for kk in [hh*8, hh*8+8) ----
        const __bf16* gsrc = wpk + ((size_t)h * 16 + hh * 8) * 8 * 32 * 16;
        for (int c = threadIdx.x; c < 4096; c += 256) {     // 4096 x 16B chunks
#if USE_ASYNC_LDS
            __builtin_amdgcn_global_load_async_to_lds_b128(
                (__attribute__((address_space(1))) v4i_ty*)(gsrc + (size_t)c * 8),
                (__attribute__((address_space(3))) v4i_ty*)&lds_b[(size_t)c * 8],
                0, 0);
#else
            *(float4*)&lds_b[(size_t)c * 8] = *(const float4*)(gsrc + (size_t)c * 8);
#endif
        }
#if USE_ASYNC_LDS
        __builtin_amdgcn_s_wait_asynccnt(0);
#endif
        __syncthreads();

        // ---- compute 8 K-steps against LDS-resident B ----
        if (active) {
            for (int kk2 = 0; kk2 < 8; ++kk2) {
                int kk = hh * 8 + kk2;
                v8bf alo = *(const v8bf*)(aptr + kk * 32);
                v8bf ahi = *(const v8bf*)(aptr + kk * 32 + 8);
                v16bf a;
#pragma unroll
                for (int i = 0; i < 8; ++i) { a[i] = alo[i]; a[8 + i] = ahi[i]; }
                // batch-load all 8 B tiles (distinct regs -> one dscnt wait)
                v16bf b[8];
#pragma unroll
                for (int t = 0; t < 8; ++t)
                    b[t] = *(const v16bf*)&lds_b[(size_t)((kk2 * 8 + t) * 32 + lane) * 16];
#pragma unroll
                for (int t = 0; t < 8; ++t)
                    acc[t] = __builtin_amdgcn_wmma_f32_16x16x32_bf16(
                        false, a, false, b[t], (short)0, acc[t], false, false);
            }
        }
        __syncthreads();   // all waves done reading LDS before restage
    }

    if (!active) return;
    // C/D layout: VGPR r -> M = r (lanes 0-15) / 8+r (lanes 16-31), N = lane&15
    int r0  = mtile * 16 + ((lane < 16) ? 0 : 8);
    int col = lane & 15;
#pragma unroll
    for (int t = 0; t < 8; ++t) {
        int o = t * 16 + col;
#pragma unroll
        for (int r = 0; r < 8; ++r) {
            int rr = r0 + r;
            if (rr < NN) xp[((size_t)rr * NH + h) * OUTD + o] = acc[t][r];
        }
    }
}

// ---------------------------------------------------------------- attention
// a_src[n][h], a_dst[n][h]: one wave per (n,h)
__global__ void k_attn_dots(const float* __restrict__ xp,
                            const float* __restrict__ att_src,
                            const float* __restrict__ att_dst,
                            float* __restrict__ a_src, float* __restrict__ a_dst) {
    int wid  = (blockIdx.x * blockDim.x + threadIdx.x) >> 5;
    int lane = threadIdx.x & 31;
    if (wid >= NN * NH) return;
    int h = wid % NH;
    const float* xr = xp + (size_t)wid * OUTD;
    const float* as = att_src + h * OUTD;
    const float* ad = att_dst + h * OUTD;
    float s0 = 0.0f, s1 = 0.0f;
#pragma unroll
    for (int i = lane; i < OUTD; i += 32) {
        float v = xr[i];
        s0 += v * as[i];
        s1 += v * ad[i];
    }
#pragma unroll
    for (int m = 16; m >= 1; m >>= 1) {
        s0 += __shfl_xor(s0, m, 32);
        s1 += __shfl_xor(s1, m, 32);
    }
    if (lane == 0) { a_src[wid] = s0; a_dst[wid] = s1; }
}

// alpha_raw = leaky_relu(a_src[src]+a_dst[dst]+a_edge); running segment max.
// One wave per edge; lane d and d+32 cover the 64 edge features.
__global__ void k_edge_scores(const float* __restrict__ edge_attr,
                              const int* __restrict__ ei,
                              const float* __restrict__ ve,
                              const float* __restrict__ a_src,
                              const float* __restrict__ a_dst,
                              float* __restrict__ alpha,
                              float* __restrict__ segmax) {
    int e    = (blockIdx.x * blockDim.x + threadIdx.x) >> 5;
    int lane = threadIdx.x & 31;
    if (e >= NE) return;
    const float* ea = edge_attr + (size_t)e * EDD;
    float e0 = ea[lane], e1 = ea[lane + 32];
    float s[NH];
#pragma unroll
    for (int h = 0; h < NH; ++h)
        s[h] = e0 * ve[h * EDD + lane] + e1 * ve[h * EDD + lane + 32];
#pragma unroll
    for (int m = 16; m >= 1; m >>= 1)
#pragma unroll
        for (int h = 0; h < NH; ++h) s[h] += __shfl_xor(s[h], m, 32);

    int src = ei[e], dst = ei[NE + e];
    if (lane < NH) {
        float ae = 0.0f;
#pragma unroll
        for (int h = 0; h < NH; ++h) if (h == lane) ae = s[h];
        float al = a_src[src * NH + lane] + a_dst[dst * NH + lane] + ae;
        al = (al >= 0.0f) ? al : NEG_SLOPE * al;        // leaky_relu
        alpha[(size_t)e * NH + lane] = al;
        atomicMaxF(&segmax[dst * NH + lane], al);
    }
}

// ex = exp(alpha - seg_max[dst]); denom[dst] += ex
__global__ void k_softmax_num(const int* __restrict__ ei,
                              float* __restrict__ alpha,
                              const float* __restrict__ segmax,
                              float* __restrict__ denom) {
    int i = blockIdx.x * blockDim.x + threadIdx.x;
    if (i >= NE * NH) return;
    int e = i / NH, h = i % NH;
    int dst = ei[NE + e];
    float ex = __expf(alpha[i] - segmax[dst * NH + h]);
    alpha[i] = ex;
    atomicAdd(&denom[dst * NH + h], ex);
}

// out[dst][h][:] += (ex/denom) * xp[src][h][:]  — one wave per (e,h), float4/lane
__global__ void k_scatter(const int* __restrict__ ei,
                          const float* __restrict__ alpha,
                          const float* __restrict__ denom,
                          const float* __restrict__ xp,
                          float* __restrict__ out) {
    int wid  = (blockIdx.x * blockDim.x + threadIdx.x) >> 5;
    int lane = threadIdx.x & 31;
    if (wid >= NE * NH) return;
    int e = wid / NH, h = wid % NH;
    int src = ei[e], dst = ei[NE + e];
    float w = alpha[wid] / (denom[dst * NH + h] + EPS_DEN);
    float4 v = ((const float4*)(xp + ((size_t)src * NH + h) * OUTD))[lane];
    float* op = out + ((size_t)dst * NH + h) * OUTD + lane * 4;
    atomicAdd(op + 0, w * v.x);
    atomicAdd(op + 1, w * v.y);
    atomicAdd(op + 2, w * v.z);
    atomicAdd(op + 3, w * v.w);
}

// ---------------------------------------------------------------- launch
extern "C" void kernel_launch(void* const* d_in, const int* in_sizes, int n_in,
                              void* d_out, int out_size, void* d_ws, size_t ws_size,
                              hipStream_t stream) {
    const float* x        = (const float*)d_in[0];
    const int*   ei       = (const int*)  d_in[1];
    const float* edge_att = (const float*)d_in[2];
    const float* W        = (const float*)d_in[3];
    const float* att_src  = (const float*)d_in[4];
    const float* att_dst  = (const float*)d_in[5];
    const float* W_edge   = (const float*)d_in[6];
    const float* att_edge = (const float*)d_in[7];
    const float* bias     = (const float*)d_in[8];
    float* out = (float*)d_out;

    // workspace carve-out (256B aligned)
    char*  ws  = (char*)d_ws;
    size_t off = 0;
    auto carve = [&](size_t bytes) {
        size_t o = off;
        off += (bytes + 255) & ~(size_t)255;
        return o;
    };
    __bf16* xb     = (__bf16*)(ws + carve((size_t)NN * IND * 2));               // 25.6 MB
    __bf16* wpk    = (__bf16*)(ws + carve((size_t)NH * 16 * 8 * 32 * 16 * 2));  // 0.66 MB
    float*  xp     = (float*) (ws + carve((size_t)NN * NH * OUTD * 4));         // 64 MB
    float*  a_src  = (float*) (ws + carve((size_t)NN * NH * 4));
    float*  a_dst  = (float*) (ws + carve((size_t)NN * NH * 4));
    float*  ve     = (float*) (ws + carve((size_t)NH * EDD * 4));
    float*  alpha  = (float*) (ws + carve((size_t)NE * NH * 4));                // 8 MB
    float*  segmax = (float*) (ws + carve((size_t)NN * NH * 4));
    float*  denom  = (float*) (ws + carve((size_t)NN * NH * 4));
    (void)in_sizes; (void)n_in; (void)out_size; (void)ws_size;

    // 1) precision/layout prep
    k_pack_x<<<(NN * IND / 4 + 255) / 256, 256, 0, stream>>>(x, xb);
    k_pack_w<<<(NH * 16 * 8 * 32 * 16 + 255) / 256, 256, 0, stream>>>(W, wpk);
    k_vedge<<<1, 512, 0, stream>>>(W_edge, att_edge, ve);
    k_init<<<(NN * NH * OUTD + 255) / 256, 256, 0, stream>>>(out, bias, denom, segmax);

    // 2) WMMA projection: xp = x @ W per head (B staged in LDS)
    k_gemm<<<NH * MB, 256, 0, stream>>>(xb, wpk, xp);

    // 3) attention logits
    k_attn_dots<<<((size_t)NN * NH * 32 + 255) / 256, 256, 0, stream>>>(
        xp, att_src, att_dst, a_src, a_dst);
    k_edge_scores<<<((size_t)NE * 32 + 255) / 256, 256, 0, stream>>>(
        edge_att, ei, ve, a_src, a_dst, alpha, segmax);

    // 4) segment softmax
    k_softmax_num<<<(NE * NH + 255) / 256, 256, 0, stream>>>(ei, alpha, segmax, denom);

    // 5) weighted scatter-aggregate (L2-resident atomics)
    k_scatter<<<((size_t)NE * NH * 32 + 255) / 256, 256, 0, stream>>>(
        ei, alpha, denom, xp, out);
}
